// kMaXTransformerLayer_30279519437263
// MI455X (gfx1250) — matrix operand order repaired and checked by reading, not compiled, not run on hardware
//
#include <hip/hip_runtime.h>
#include <hip/hip_bf16.h>
#include <math.h>

// ---------------------------------------------------------------------------
// kMaX transformer layer for MI455X (gfx1250, wave32).
// All 1x1 convs run as bf16 WMMA GEMMs (v_wmma_f32_16x16x32_bf16, f32 accum).
// Activations are GELU'd + converted to bf16 once, halving GEMM read traffic
// (pixel_feature alone is 67MB; HBM 23.3TB/s is the floor for this layer).
// GEMM: 128(M)x64(N) block tile, A tile staged to LDS with
// GLOBAL_LOAD_ASYNC_TO_LDS_B128 (ASYNCcnt path), B tile staged transposed.
// BatchNorm (training mode) = per-channel reduce + fused apply(+GELU)(+bf16).
// k-means aggregation = argmax + float atomics (cheaper than dense one-hot GEMM).
// ---------------------------------------------------------------------------

typedef __attribute__((ext_vector_type(16))) __bf16 v16bf;
typedef __attribute__((ext_vector_type(8)))  __bf16 v8bf;
typedef __attribute__((ext_vector_type(8)))  float  v8f;

#define BN_EPS 1e-5f

__device__ __forceinline__ float gelu_f(float x) {
  return 0.5f * x * (1.0f + erff(x * 0.70710678118654752440f)); // exact erf GELU
}
__device__ __forceinline__ __bf16 f2bf(float x) { return (__bf16)x; }

// ---------------------------------------------------------------------------
// Batched GEMM: C[z][M,Nn] = A[z][M,K] * B[z][K,Nn]   (row-major, bf16 in, f32 out)
// Block = 128 threads (4 waves) computing a 128(M) x 64(N) tile; wave w owns
// rows {w*16..w*16+15} and {64+w*16..}. Per 32-K chunk: A[128x32] is staged to
// LDS via async b128 copies (contiguous, lane-addressed, ASYNCcnt-tracked),
// B[32x64] is staged transposed so B fragments are two ds_load_b128 per lane.
// Fragment layouts per CDNA5 ISA 7.12.2:
//   A (16-bit 16x32): lanes 0-15 / 16-31 hold K halves (half = lane>>4),
//     per lane two contiguous 8-element runs (K = half*8.. and 16+half*8..).
//   C/D (f32 16x16): VGPR r -> M = r + half*8, N = lane&15.
// ---------------------------------------------------------------------------
__global__ __launch_bounds__(128)
void gemm_bf16_wmma(const __bf16* __restrict__ A, const __bf16* __restrict__ B,
                    float* __restrict__ C, int M, int Nn, int K,
                    long aStride, long bStride, long cStride) {
  __shared__ __bf16 At[128 * 40]; // [row][k], row stride 80B (16B-aligned, bank-spread)
  __shared__ __bf16 Bt[64][40];   // [n][k] transposed tile, row stride 80B
  const int lane = threadIdx.x;           // 0..31
  const int wave = threadIdx.y;           // 0..3
  const int t    = wave * 32 + lane;      // 0..127
  const int z    = blockIdx.z;
  A += z * aStride; B += z * bStride; C += z * cStride;
  const int mBlock  = blockIdx.y * 128;
  const int colBase = blockIdx.x * 64;
  const int half = lane >> 4;
  const int m16  = lane & 15;

  v8f zero = {0.f, 0.f, 0.f, 0.f, 0.f, 0.f, 0.f, 0.f};
  v8f acc[2][4];
#pragma unroll
  for (int mt = 0; mt < 2; ++mt)
#pragma unroll
    for (int nt = 0; nt < 4; ++nt) acc[mt][nt] = zero;

  for (int k0 = 0; k0 < K; k0 += 32) {
    // --- async-stage A[128 x 32] -> At (contiguous 16B chunks, ASYNCcnt) ---
#pragma unroll
    for (int i = 0; i < 4; ++i) {
      const int c   = t + 128 * i;     // 512 chunks of 16B = 8KB tile
      const int row = c >> 2, q = c & 3;
      const __bf16* src = A + (long)(mBlock + row) * K + k0 + q * 8;
      const unsigned dst = (unsigned)(size_t)&At[row * 40 + q * 8]; // LDS offset
      asm volatile("global_load_async_to_lds_b128 %0, %1, off"
                   :: "v"(dst), "v"(src) : "memory");
    }
    // --- stage B[32 x 64] -> Bt[64][32] (transposed), coalesced 16B loads ---
    int g = t;
#pragma unroll
    for (int it = 0; it < 2; ++it) {
      const int kr = g >> 3;   // 0..31
      const int cg = g & 7;    // 0..7 (8 cols each)
      v8bf v = *(const v8bf*)(B + (long)(k0 + kr) * Nn + colBase + cg * 8);
#pragma unroll
      for (int i = 0; i < 8; ++i) Bt[cg * 8 + i][kr] = v[i];
      g += 128;
    }
    // prefetch next B K-chunk into cache (global_prefetch_b8)
    if (k0 + 32 < K)
      __builtin_prefetch(B + (long)(k0 + 32 + (t >> 2)) * Nn + colBase + (t & 3) * 16, 0, 1);
    asm volatile("s_wait_asynccnt 0x0" ::: "memory"); // this wave's async copies done
    __syncthreads();

    // --- B fragments (shared across both M tiles) ---
    v16bf bfrag[4];
#pragma unroll
    for (int nt = 0; nt < 4; ++nt) {
      const int cn = nt * 16 + m16;
      const int kb = half * 8;
      v8bf b_lo = *(const v8bf*)(&Bt[cn][kb]);
      v8bf b_hi = *(const v8bf*)(&Bt[cn][kb + 16]);
#pragma unroll
      for (int i = 0; i < 8; ++i) { bfrag[nt][i] = b_lo[i]; bfrag[nt][8 + i] = b_hi[i]; }
    }
    // --- A fragments from LDS ---
    v16bf afrag[2];
#pragma unroll
    for (int mt = 0; mt < 2; ++mt) {
      const int row  = mt * 64 + wave * 16 + m16;
      const int base = row * 40 + half * 8;
      v8bf a_lo = *(const v8bf*)(&At[base]);
      v8bf a_hi = *(const v8bf*)(&At[base + 16]);
#pragma unroll
      for (int i = 0; i < 8; ++i) { afrag[mt][i] = a_lo[i]; afrag[mt][8 + i] = a_hi[i]; }
    }
#pragma unroll
    for (int mt = 0; mt < 2; ++mt)
#pragma unroll
      for (int nt = 0; nt < 4; ++nt)
        acc[mt][nt] = __builtin_amdgcn_wmma_f32_16x16x32_bf16(
            false, afrag[mt], false, bfrag[nt], (short)0, acc[mt][nt], false, false);
    __syncthreads();
  }

#pragma unroll
  for (int mt = 0; mt < 2; ++mt)
#pragma unroll
    for (int nt = 0; nt < 4; ++nt) {
      const int col = colBase + nt * 16 + m16;
#pragma unroll
      for (int r = 0; r < 8; ++r)
        C[(long)(mBlock + mt * 64 + wave * 16 + r + half * 8) * Nn + col] = acc[mt][nt][r];
    }
}

// ---------------------------------------------------------------------------
// float -> bf16 (optionally through exact GELU)
// ---------------------------------------------------------------------------
__global__ void cvt_bf16_k(const float* __restrict__ x, __bf16* __restrict__ y,
                           long n, int doGelu) {
  for (long i = blockIdx.x * 256L + threadIdx.x; i < n; i += (long)gridDim.x * 256) {
    float v = x[i];
    if (doGelu) v = gelu_f(v);
    y[i] = f2bf(v);
  }
}

// ---------------------------------------------------------------------------
// per-channel mean / population variance. x addressed as x[b*bStride + c*chStride + s]
// ---------------------------------------------------------------------------
__global__ __launch_bounds__(256)
void reduce_meanvar(const float* __restrict__ x, float* __restrict__ mean,
                    float* __restrict__ var, int NB, int S,
                    long chStride, long bStride, float invCnt) {
  const int c = blockIdx.x;
  const float* xc = x + (long)c * chStride;
  float s = 0.f, s2 = 0.f;
  for (int b = 0; b < NB; ++b) {
    const float* xb = xc + (long)b * bStride;
    for (int i = threadIdx.x; i < S; i += 256) {
      float v = xb[i];
      s += v; s2 += v * v;
    }
  }
  __shared__ float sh[256], sh2[256];
  sh[threadIdx.x] = s; sh2[threadIdx.x] = s2;
  __syncthreads();
  for (int o = 128; o > 0; o >>= 1) {
    if (threadIdx.x < o) { sh[threadIdx.x] += sh[threadIdx.x + o]; sh2[threadIdx.x] += sh2[threadIdx.x + o]; }
    __syncthreads();
  }
  if (threadIdx.x == 0) {
    float m = sh[0] * invCnt;
    mean[c] = m;
    var[c] = sh2[0] * invCnt - m * m;
  }
}

// ---------------------------------------------------------------------------
// y = (x - mean[c]) * rsqrt(var[c]+eps) * g[c] + b[c]  [+ GELU]  -> f32 and/or bf16
// layout [NB][C][S] contiguous
// ---------------------------------------------------------------------------
__global__ void bn_apply(const float* __restrict__ x, const float* __restrict__ mean,
                         const float* __restrict__ var, const float* __restrict__ g,
                         const float* __restrict__ b, float* __restrict__ outf,
                         __bf16* __restrict__ outbf, int C, long S, long total,
                         int doGelu) {
  for (long i = blockIdx.x * 256L + threadIdx.x; i < total; i += (long)gridDim.x * 256) {
    int c = (int)((i / S) % C);
    float y = (x[i] - mean[c]) * rsqrtf(var[c] + BN_EPS) * g[c] + b[c];
    if (doGelu) y = gelu_f(y);
    if (outf)  outf[i] = y;
    if (outbf) outbf[i] = f2bf(y);
  }
}

// depthwise 5x5 conv, pad 2; x,y: [NB][256][64][64], w: [256][25]
__global__ void dwconv5x5_k(const float* __restrict__ x, const float* __restrict__ w,
                            float* __restrict__ y, long total) {
  for (long i = blockIdx.x * 256L + threadIdx.x; i < total; i += (long)gridDim.x * 256) {
    int px = (int)(i & 63), py = (int)((i >> 6) & 63), c = (int)((i >> 12) & 255);
    const float* xc = x + (i & ~4095L);          // base of this [n][c] plane
    const float* wc = w + c * 25;
    float s = 0.f;
#pragma unroll
    for (int ky = 0; ky < 5; ++ky) {
      int yy = py + ky - 2;
      if (yy < 0 || yy > 63) continue;
#pragma unroll
      for (int kx = 0; kx < 5; ++kx) {
        int xx = px + kx - 2;
        if (xx < 0 || xx > 63) continue;
        s += xc[yy * 64 + xx] * wc[ky * 5 + kx];
      }
    }
    y[i] = s;
  }
}

// per-pixel L2 normalize over 128 channels; x:[NB][128][4096]
__global__ void l2norm128_k(const float* __restrict__ x, float* __restrict__ outf,
                            __bf16* __restrict__ outbf, long total) {
  for (long i = blockIdx.x * 256L + threadIdx.x; i < total; i += (long)gridDim.x * 256) {
    long n = i >> 12; long p = i & 4095;
    const float* col = x + n * 128 * 4096 + p;
    float ss = 0.f;
    for (int c = 0; c < 128; ++c) { float v = col[c * 4096]; ss += v * v; }
    float inv = 1.0f / fmaxf(sqrtf(ss), 1e-12f);
    for (int c = 0; c < 128; ++c) {
      float v = col[c * 4096] * inv;
      long o = n * 128 * 4096 + (long)c * 4096 + p;
      outf[o] = v; outbf[o] = f2bf(v);
    }
  }
}

// class logits: out[n][l][j] = sum_c qs[n][c][l]*W[j][c] + b[j] (+void bias on j=132)
__global__ void class_logits_k(const float* __restrict__ qs, const float* __restrict__ Wc,
                               const float* __restrict__ bc, float* __restrict__ out,
                               long total) {
  const float voidBias = logf(1188.0f); // log((133-1)*0.9/0.1)
  for (long i = blockIdx.x * 256L + threadIdx.x; i < total; i += (long)gridDim.x * 256) {
    int j = (int)(i % 133); long r = i / 133;
    int l = (int)(r & 127); long n = r >> 7;
    const float* q = qs + (n * 256) * 128 + l;
    const float* w = Wc + j * 256;
    float s = bc[j];
    for (int c = 0; c < 256; ++c) s += q[c * 128] * w[c];
    if (j == 132) s += voidBias;
    out[i] = s;
  }
}

// mkT[n][l][c] = (bf16) mk[n][c][l]   (128x128 per batch)
__global__ void transpose_cvt_k(const float* __restrict__ mk, __bf16* __restrict__ out,
                                long total) {
  for (long i = blockIdx.x * 256L + threadIdx.x; i < total; i += (long)gridDim.x * 256) {
    long n = i >> 14; long r = i & 16383;
    int l = (int)(r >> 7), c = (int)(r & 127);
    out[i] = f2bf(mk[(n << 14) + ((long)c << 7) + l]);
  }
}

// argmax over L=128 of mask logits [NB][128][4096] -> assign[NB][4096]
__global__ void argmax_l_k(const float* __restrict__ ml, int* __restrict__ assign,
                           long total) {
  for (long i = blockIdx.x * 256L + threadIdx.x; i < total; i += (long)gridDim.x * 256) {
    long n = i >> 12; long p = i & 4095;
    const float* col = ml + n * 128 * 4096 + p;
    float best = col[0]; int bl = 0;
    for (int l = 1; l < 128; ++l) {
      float v = col[l * 4096];
      if (v > best) { best = v; bl = l; }   // first max (np/jnp semantics)
    }
    assign[i] = bl;
  }
}

// km[n][c][assign[n][p]] += pv[n][c][p]
__global__ void kmeans_scatter_k(const float* __restrict__ pv, const int* __restrict__ assign,
                                 float* __restrict__ km, long total) {
  for (long i = blockIdx.x * 256L + threadIdx.x; i < total; i += (long)gridDim.x * 256) {
    long p = i & 4095; long c = (i >> 12) & 255; long n = i >> 20;
    int l = assign[(n << 12) + p];
    atomicAdd(&km[(n * 256 + c) * 128 + l], pv[i]);
  }
}

// sim[n][h][l][m] = sum_{d<16} q[n][h*16+d][l] * k[n][128+h*16+d][m]  (qkv:[NB][512][128])
__global__ void attn_sim_k(const float* __restrict__ qkv, float* __restrict__ sim,
                           long total) {
  for (long i = blockIdx.x * 256L + threadIdx.x; i < total; i += (long)gridDim.x * 256) {
    int m = (int)(i & 127), l = (int)((i >> 7) & 127), h = (int)((i >> 14) & 7);
    long n = i >> 17;
    const float* q = qkv + (n * 512 + h * 16) * 128 + l;
    const float* k = qkv + (n * 512 + 128 + h * 16) * 128 + m;
    float s = 0.f;
#pragma unroll
    for (int d = 0; d < 16; ++d) s += q[d * 128] * k[d * 128];
    sim[i] = s;
  }
}

// row softmax over last dim (128)
__global__ void softmax128_k(float* __restrict__ sim, long rows) {
  for (long i = blockIdx.x * 256L + threadIdx.x; i < rows; i += (long)gridDim.x * 256) {
    float* r = sim + i * 128;
    float mx = r[0];
    for (int m = 1; m < 128; ++m) mx = fmaxf(mx, r[m]);
    float s = 0.f;
    for (int m = 0; m < 128; ++m) { float e = __expf(r[m] - mx); r[m] = e; s += e; }
    float inv = 1.0f / s;
    for (int m = 0; m < 128; ++m) r[m] *= inv;
  }
}

// rv[n][c][l] = sum_m attn[n][h][l][m] * v[n][256+h*32+d][m], c = h*32+d
__global__ void attn_rv_k(const float* __restrict__ attn, const float* __restrict__ qkv,
                          float* __restrict__ rv, long total) {
  for (long i = blockIdx.x * 256L + threadIdx.x; i < total; i += (long)gridDim.x * 256) {
    int l = (int)(i & 127), c = (int)((i >> 7) & 255);
    long n = i >> 15;
    int h = c >> 5;
    const float* a = attn + ((n * 8 + h) * 128 + l) * 128;
    const float* v = qkv + (n * 512 + 256 + c) * 128;
    float s = 0.f;
    for (int m = 0; m < 128; ++m) s += a[m] * v[m];
    rv[i] = s;
  }
}

// out = a + b  [+ GELU]  -> f32 and/or bf16
__global__ void add2_k(const float* __restrict__ a, const float* __restrict__ b,
                       float* __restrict__ outf, __bf16* __restrict__ outbf,
                       long n, int doGelu) {
  for (long i = blockIdx.x * 256L + threadIdx.x; i < n; i += (long)gridDim.x * 256) {
    float y = a[i] + b[i];
    if (doGelu) y = gelu_f(y);
    if (outf)  outf[i] = y;
    if (outbf) outbf[i] = f2bf(y);
  }
}

// ---------------------------------------------------------------------------
static inline int nblk(long n) {
  long b = (n + 255) / 256;
  return (int)(b > 8192 ? 8192 : b);
}

extern "C" void kernel_launch(void* const* d_in, const int* in_sizes, int n_in,
                              void* d_out, int out_size, void* d_ws, size_t ws_size,
                              hipStream_t stream) {
  // dims
  const long HW = 4096;                 // 64*64
  const long PIX = 2L * 256 * HW;       // 2,097,152
  const long PIX128 = 2L * 128 * HW;
  const long QEL = 2L * 256 * 128;      // 65,536

  // inputs (setup_inputs dict order, params flattened in insertion order)
  const float* pf   = (const float*)d_in[0];
  const float* qfeat= (const float*)d_in[1];
  const float* Wp1 = (const float*)d_in[2],  *gp1 = (const float*)d_in[3],  *bp1 = (const float*)d_in[4];
  const float* Wq1 = (const float*)d_in[5],  *gq1 = (const float*)d_in[6],  *bq1 = (const float*)d_in[7];
  const float* Wpv = (const float*)d_in[8],  *gpv = (const float*)d_in[9],  *bpv = (const float*)d_in[10];
  const float* Wdw = (const float*)d_in[11], *gdw = (const float*)d_in[12], *bdw = (const float*)d_in[13];
  const float* Wpc1= (const float*)d_in[14], *gpc1= (const float*)d_in[15], *bpc1= (const float*)d_in[16];
  const float* Wlast=(const float*)d_in[17]; /* blast (d_in[18]) cancels inside BN */
  const float* glast=(const float*)d_in[19], *bnlast=(const float*)d_in[20];
  const float* Wcls =(const float*)d_in[21], *bcls=(const float*)d_in[22];
  const float* Wmsk =(const float*)d_in[23], *gmsk=(const float*)d_in[24], *bmsk=(const float*)d_in[25];
  const float* gpm  =(const float*)d_in[26], *bpm =(const float*)d_in[27];
  const float* gkm  =(const float*)d_in[28], *bkm =(const float*)d_in[29];
  const float* Wkm3 =(const float*)d_in[30], *gkm3=(const float*)d_in[31], *bkm3=(const float*)d_in[32];
  const float* Wqkv =(const float*)d_in[33], *gqkv=(const float*)d_in[34], *bqkv=(const float*)d_in[35];
  const float* gsim =(const float*)d_in[36], *bsim=(const float*)d_in[37];
  const float* grv  =(const float*)d_in[38], *brv =(const float*)d_in[39];
  const float* Wq3  =(const float*)d_in[40], *gq3 =(const float*)d_in[41], *bq3 =(const float*)d_in[42];
  const float* Wf1  =(const float*)d_in[43], *gf1 =(const float*)d_in[44], *bf1 =(const float*)d_in[45];
  const float* Wf2  =(const float*)d_in[46], *gf2 =(const float*)d_in[47], *bf2 =(const float*)d_in[48];

  // d_out sections (return order: qf, class_logits, mask_logits, pix_norm)
  float* out_qf   = (float*)d_out;                       // 65,536
  float* out_cls  = out_qf + 65536;                      // 34,048
  float* out_mask = out_cls + 34048;                     // 1,048,576
  float* out_pn   = out_mask + 1048576;                  // 1,048,576

  // workspace bump allocator (256B aligned)
  char* wsp = (char*)d_ws;
  auto alloc = [&](size_t bytes) -> void* {
    void* p = wsp; wsp += (bytes + 255) & ~(size_t)255; return p;
  };
  __bf16* bf_pf   = (__bf16*)alloc(2L*2048*HW*2);
  __bf16* bfWp1   = (__bf16*)alloc(256L*2048*2);
  __bf16* bfWq1   = (__bf16*)alloc(256L*256*2);
  __bf16* bfWpv   = (__bf16*)alloc(256L*256*2);
  __bf16* bfWpc1  = (__bf16*)alloc(256L*256*2);
  __bf16* bfWlast = (__bf16*)alloc(128L*256*2);
  __bf16* bfWmsk  = (__bf16*)alloc(128L*256*2);
  __bf16* bfWkm3  = (__bf16*)alloc(256L*256*2);
  __bf16* bfWqkv  = (__bf16*)alloc(512L*256*2);
  __bf16* bfWq3   = (__bf16*)alloc(256L*256*2);
  __bf16* bfWf1   = (__bf16*)alloc(2048L*256*2);
  __bf16* bfWf2   = (__bf16*)alloc(256L*2048*2);
  __bf16* bf_qf   = (__bf16*)alloc(QEL*2);
  float*  meanp   = (float*)alloc(2048*4);
  float*  varp    = (float*)alloc(2048*4);
  float*  graw    = (float*)alloc(PIX*4);      // pixel GEMM scratch
  float*  ps_f32  = (float*)alloc(PIX*4);      // pixel_space
  __bf16* ps_bf   = (__bf16*)alloc(PIX*2);
  float*  pv_f32  = (float*)alloc(PIX*4);      // pixel_value (persists for kmeans)
  float*  dw_f32  = (float*)alloc(PIX*4);
  __bf16* t1_bf   = (__bf16*)alloc(PIX*2);
  float*  ps128   = (float*)alloc(PIX128*4);
  __bf16* pn_bf   = (__bf16*)alloc(PIX128*2);
  float*  qs_raw  = (float*)alloc(QEL*4);      // reused scratch for small GEMMs
  float*  qs_f32  = (float*)alloc(QEL*4);
  __bf16* qs_bf   = (__bf16*)alloc(QEL*2);
  float*  mk_f32  = (float*)alloc(2L*128*128*4);
  __bf16* mkT_bf  = (__bf16*)alloc(2L*128*128*2);
  int*    assign  = (int*)alloc(2L*4096*4);
  float*  km_f32  = (float*)alloc(QEL*4);
  __bf16* km_bf   = (__bf16*)alloc(QEL*2);
  float*  km3_raw = (float*)alloc(QEL*4);
  float*  qf_f32  = (float*)alloc(QEL*4);
  float*  qkv_raw = (float*)alloc(2L*512*128*4);
  float*  qkv_f32 = (float*)alloc(2L*512*128*4);
  float*  sim     = (float*)alloc(2L*8*128*128*4);
  float*  rv_f32  = (float*)alloc(QEL*4);
  __bf16* rv_bf   = (__bf16*)alloc(QEL*2);
  float*  q3_raw  = (float*)alloc(QEL*4);
  float*  qf2_f32 = (float*)alloc(QEL*4);
  __bf16* qf2_bf  = (__bf16*)alloc(QEL*2);
  float*  ffn1_raw= (float*)alloc(2L*2048*128*4);
  __bf16* ffn1_bf = (__bf16*)alloc(2L*2048*128*2);
  float*  ffn2_raw= (float*)alloc(QEL*4);

  auto gemm = [&](const __bf16* A, const __bf16* B, float* Cp, int M, int Nn, int K,
                  long aS, long bS, long cS) {
    dim3 g(Nn / 64, M / 128, 2), blk(32, 4);
    gemm_bf16_wmma<<<g, blk, 0, stream>>>(A, B, Cp, M, Nn, K, aS, bS, cS);
  };
  auto stats = [&](const float* x, int Cc, int S, long chS, long bS) {
    reduce_meanvar<<<dim3(Cc), dim3(256), 0, stream>>>(x, meanp, varp, 2, S, chS, bS,
                                                       1.0f / (2.0f * S));
  };
  auto bnap = [&](const float* x, const float* g, const float* b, float* of, __bf16* ob,
                  int Cc, long S, int gl) {
    long total = 2L * Cc * S;
    bn_apply<<<nblk(total), 256, 0, stream>>>(x, meanp, varp, g, b, of, ob, Cc, S, total, gl);
  };
  auto cvt = [&](const float* x, __bf16* y, long n, int gl) {
    cvt_bf16_k<<<nblk(n), 256, 0, stream>>>(x, y, n, gl);
  };

  // ---- prepack: gelu(pixel_feature)->bf16, weights->bf16, query_feature->bf16 ----
  cvt(pf, bf_pf, 2L * 2048 * HW, 1);
  cvt(Wp1, bfWp1, 256L * 2048, 0);   cvt(Wq1, bfWq1, 256L * 256, 0);
  cvt(Wpv, bfWpv, 256L * 256, 0);    cvt(Wpc1, bfWpc1, 256L * 256, 0);
  cvt(Wlast, bfWlast, 128L * 256, 0);cvt(Wmsk, bfWmsk, 128L * 256, 0);
  cvt(Wkm3, bfWkm3, 256L * 256, 0);  cvt(Wqkv, bfWqkv, 512L * 256, 0);
  cvt(Wq3, bfWq3, 256L * 256, 0);    cvt(Wf1, bfWf1, 2048L * 256, 0);
  cvt(Wf2, bfWf2, 256L * 2048, 0);   cvt(qfeat, bf_qf, QEL, 0);

  // ---- pixel_space = gelu(bn(Wp1 @ gelu(pf))) ----
  gemm(bfWp1, bf_pf, graw, 256, 4096, 2048, 0, 2048L * HW, 256L * HW);
  stats(graw, 256, 4096, HW, 256L * HW);
  bnap(graw, gp1, bp1, ps_f32, ps_bf, 256, HW, 1);

  // ---- pixel_value = bn(Wpv @ pixel_space) ----
  gemm(bfWpv, ps_bf, graw, 256, 4096, 256, 0, 256L * HW, 256L * HW);
  stats(graw, 256, 4096, HW, 256L * HW);
  bnap(graw, gpv, bpv, pv_f32, nullptr, 256, HW, 0);

  // ---- predictor: dw5x5 -> bn+gelu -> 1x1 -> bn+gelu -> 1x1(128) -> bn ----
  dwconv5x5_k<<<nblk(PIX), 256, 0, stream>>>(ps_f32, Wdw, dw_f32, PIX);
  stats(dw_f32, 256, 4096, HW, 256L * HW);
  bnap(dw_f32, gdw, bdw, nullptr, t1_bf, 256, HW, 1);
  gemm(bfWpc1, t1_bf, graw, 256, 4096, 256, 0, 256L * HW, 256L * HW);
  stats(graw, 256, 4096, HW, 256L * HW);
  bnap(graw, gpc1, bpc1, nullptr, t1_bf, 256, HW, 1);
  gemm(bfWlast, t1_bf, ps128, 128, 4096, 256, 0, 256L * HW, 128L * HW);
  stats(ps128, 128, 4096, HW, 128L * HW);                 // +blast cancels in BN
  bnap(ps128, glast, bnlast, ps128, nullptr, 128, HW, 0);
  l2norm128_k<<<nblk(2L * HW), 256, 0, stream>>>(ps128, out_pn, pn_bf, 2L * HW);

  // ---- query_space = gelu(bn(Wq1 @ query_feature)) ----
  gemm(bfWq1, bf_qf, qs_raw, 256, 128, 256, 0, 256L * 128, 256L * 128);
  stats(qs_raw, 256, 128, 128, 256L * 128);
  bnap(qs_raw, gq1, bq1, qs_f32, qs_bf, 256, 128, 1);

  // ---- class logits (incl. void bias) ----
  class_logits_k<<<nblk(2L * 128 * 133), 256, 0, stream>>>(qs_f32, Wcls, bcls, out_cls,
                                                           2L * 128 * 133);

  // ---- mask kernel + mask logits + BN(1ch) ----
  gemm(bfWmsk, qs_bf, qs_raw, 128, 128, 256, 0, 256L * 128, 128L * 128);
  stats(qs_raw, 128, 128, 128, 128L * 128);
  bnap(qs_raw, gmsk, bmsk, mk_f32, nullptr, 128, 128, 0);
  transpose_cvt_k<<<nblk(2L * 128 * 128), 256, 0, stream>>>(mk_f32, mkT_bf, 2L * 128 * 128);
  gemm(mkT_bf, pn_bf, out_mask, 128, 4096, 128, 128L * 128, 128L * HW, 128L * HW);
  stats(out_mask, 1, 128 * 4096, 0, 128L * HW);
  bnap(out_mask, gpm, bpm, out_mask, nullptr, 1, 128L * HW, 0);

  // ---- k-means: argmax (BN is monotone, matches reference order) + scatter ----
  argmax_l_k<<<nblk(2L * HW), 256, 0, stream>>>(out_mask, assign, 2L * HW);
  hipMemsetAsync(km_f32, 0, QEL * 4, stream);
  kmeans_scatter_k<<<nblk(PIX), 256, 0, stream>>>(pv_f32, assign, km_f32, PIX);
  stats(km_f32, 256, 128, 128, 256L * 128);
  bnap(km_f32, gkm, bkm, km_f32, km_bf, 256, 128, 0);
  gemm(bfWkm3, km_bf, km3_raw, 256, 128, 256, 0, 256L * 128, 256L * 128);
  stats(km3_raw, 256, 128, 128, 256L * 128);
  bnap(km3_raw, gkm3, bkm3, km3_raw, nullptr, 256, 128, 0);
  add2_k<<<nblk(QEL), 256, 0, stream>>>(qfeat, km3_raw, qf_f32, nullptr, QEL, 0);

  // ---- self-attention ----
  gemm(bfWqkv, qs_bf, qkv_raw, 512, 128, 256, 0, 256L * 128, 512L * 128);
  stats(qkv_raw, 512, 128, 128, 512L * 128);
  bnap(qkv_raw, gqkv, bqkv, qkv_f32, nullptr, 512, 128, 0);
  attn_sim_k<<<nblk(2L * 8 * 128 * 128), 256, 0, stream>>>(qkv_f32, sim, 2L * 8 * 128 * 128);
  stats(sim, 8, 128 * 128, 128L * 128, 8L * 128 * 128);
  bnap(sim, gsim, bsim, sim, nullptr, 8, 128L * 128, 0);
  softmax128_k<<<nblk(2L * 8 * 128), 256, 0, stream>>>(sim, 2L * 8 * 128);
  attn_rv_k<<<nblk(QEL), 256, 0, stream>>>(sim, qkv_f32, rv_f32, QEL);
  stats(rv_f32, 256, 128, 128, 256L * 128);
  bnap(rv_f32, grv, brv, nullptr, rv_bf, 256, 128, 1);
  gemm(bfWq3, rv_bf, q3_raw, 256, 128, 256, 0, 256L * 128, 256L * 128);
  stats(q3_raw, 256, 128, 128, 256L * 128);
  bnap(q3_raw, gq3, bq3, q3_raw, nullptr, 256, 128, 0);
  add2_k<<<nblk(QEL), 256, 0, stream>>>(qf_f32, q3_raw, qf2_f32, qf2_bf, QEL, 1);

  // ---- FFN ----
  gemm(bfWf1, qf2_bf, ffn1_raw, 2048, 128, 256, 0, 256L * 128, 2048L * 128);
  stats(ffn1_raw, 2048, 128, 128, 2048L * 128);
  bnap(ffn1_raw, gf1, bf1, nullptr, ffn1_bf, 2048, 128, 1);
  gemm(bfWf2, ffn1_bf, ffn2_raw, 256, 128, 2048, 0, 2048L * 128, 256L * 128);
  stats(ffn2_raw, 256, 128, 128, 256L * 128);
  bnap(ffn2_raw, gf2, bf2, ffn2_raw, nullptr, 256, 128, 0);
  add2_k<<<nblk(QEL), 256, 0, stream>>>(qf2_f32, ffn2_raw, out_qf, nullptr, QEL, 1);

  (void)in_sizes; (void)n_in; (void)out_size; (void)ws_size;
}
